// DTIHarmonicIS_76038101008555
// MI455X (gfx1250) — matrix operand order repaired and checked by reading, not compiled, not run on hardware
//
#include <hip/hip_runtime.h>
#include <hip/hip_bf16.h>

// ---------------------------------------------------------------------------
// DTIHarmonic for MI455X (gfx1250, wave32, WMMA).
// - hpair@W1 factorized into h1e@W1_top + h2e@W1_bot (240 GFLOP -> ~0.5 GFLOP)
// - all GEMMs use bf16 WMMA (v_wmma_f32_16x16x32_bf16), K % 32 == 0,
//   B operands pre-transposed to [N][K] so every fragment load is a b128.
// - each wave computes a 16x64 strip: A fragment loaded once per K-step; all
//   4 B fragments staged into distinct registers, then 4 back-to-back WMMAs
//   (no WAR overlap between WMMA sources and the next subtile's converts).
// - sparse energy pass gated on A_int (5% density), LDS-cached rows.
// ---------------------------------------------------------------------------

typedef __attribute__((ext_vector_type(16))) __bf16 v16bf;
typedef __attribute__((ext_vector_type(8)))  float  v8f;

#define DTI_B   8
#define DTI_N1  64
#define DTI_N2  512
#define DTI_D   128
#define DTI_H   128
#define DTI_NT  7

__constant__ float c_BCON[7] = {1.159f, 0.448f, 0.927f, 0.902f, 0.349f, 0.789f, 0.198f};

__device__ __forceinline__ void dti_put4(v16bf& f, int i, float4 v)
{
    f[i]     = (__bf16)v.x;
    f[i + 1] = (__bf16)v.y;
    f[i + 2] = (__bf16)v.z;
    f[i + 3] = (__bf16)v.w;
}

// ---------------------------------------------------------------------------
// bf16 WMMA GEMM: C[z] = act( A[z] * BT[z]^T + bias[z] ).
// A: [M][K] f32 (lda), BT: [N][K] f32 (ldbt), C: [M][N] f32 (ldc).
// Requirements: K % 32 == 0; M multiple of 16; N multiple of 64; 16B rows.
// One wave = one 16x64 strip (4 accumulators). Fragment layouts this code uses (contiguous k, not ISA table order):
//   A (16x32): lane<16 -> M=lane; K = s + 16*half            => one run of 16.
//   B (32x16): N = lane&15;  K = s + 16*half                 => one run of 16.
//   C: row = r + 8*half, col = lane&15.
// ---------------------------------------------------------------------------
__global__ __launch_bounds__(32)
void dti_gemm_wmma(const float* __restrict__ A, const float* __restrict__ BT,
                   float* __restrict__ Cm, const float* __restrict__ bias,
                   int K, int lda, int ldbt, int ldc, int doRelu,
                   long long sAz, long long sBz, long long sCz, long long sBiasz)
{
    const int z = blockIdx.z;
    A  += (long long)z * sAz;
    BT += (long long)z * sBz;
    Cm += (long long)z * sCz;
    if (bias) bias += (long long)z * sBiasz;

    const int lane = threadIdx.x & 31;
    const int half = lane >> 4;
    const int l16  = lane & 15;

    const float* arow  = A  + (long long)(blockIdx.x * 16 + l16) * lda;
    const float* brow0 = BT + (long long)(blockIdx.y * 64 + l16) * ldbt;

    v8f acc[4] = {};
    for (int k0 = 0; k0 < K; k0 += 32) {
        // A fragment: one run of 16 consecutive K values.
        const float4* ap0 = (const float4*)(arow + k0 + 16 * half);
        float4 a0 = ap0[0], a1 = ap0[1];
        float4 a2 = ap0[2], a3 = ap0[3];
        v16bf af;
        dti_put4(af, 0, a0);  dti_put4(af, 4, a1);
        dti_put4(af, 8, a2);  dti_put4(af, 12, a3);

        // Stage all 4 B fragments in distinct registers (loads all in flight),
        v16bf bf_[4];
#pragma unroll
        for (int nt = 0; nt < 4; ++nt) {
            const float4* bp = (const float4*)(brow0 + (long long)nt * 16 * ldbt
                                               + k0 + 16 * half);
            float4 b0 = bp[0], b1 = bp[1], b2 = bp[2], b3 = bp[3];
            dti_put4(bf_[nt], 0, b0); dti_put4(bf_[nt], 4, b1);
            dti_put4(bf_[nt], 8, b2); dti_put4(bf_[nt], 12, b3);
        }
        // ...then issue 4 back-to-back independent WMMAs.
#pragma unroll
        for (int nt = 0; nt < 4; ++nt)
            acc[nt] = __builtin_amdgcn_wmma_f32_16x16x32_bf16(
                          false, af, false, bf_[nt], (short)0, acc[nt], false, false);
    }

#pragma unroll
    for (int nt = 0; nt < 4; ++nt) {
        const int col = blockIdx.y * 64 + nt * 16 + l16;
        const float bv = bias ? bias[col] : 0.0f;
#pragma unroll
        for (int r = 0; r < 8; ++r) {
            const int row = blockIdx.x * 16 + r + 8 * half;
            float v = acc[nt][r] + bv;
            if (doRelu) v = fmaxf(v, 0.0f);
            Cm[(long long)row * ldc + col] = v;
        }
    }
}

// ---------------------------------------------------------------------------
// Batched LDS-tiled transpose with zero row-padding:
//   dst[z][c][r] = (r < rows) ? src[z][r][c] : 0,   for r in [0, ldd).
// grid = (ceil(ldd/16), ceil(cols/16), Z), block = 256.
// ---------------------------------------------------------------------------
__global__ __launch_bounds__(256)
void dti_transpose(const float* __restrict__ src, float* __restrict__ dst,
                   int rows, int cols, int ldd,
                   long long sSrcz, long long sDstz)
{
    __shared__ float tile[16][17];
    src += (long long)blockIdx.z * sSrcz;
    dst += (long long)blockIdx.z * sDstz;
    const int tx = threadIdx.x & 15, ty = threadIdx.x >> 4;
    const int r0 = blockIdx.x * 16, c0 = blockIdx.y * 16;
    const int r = r0 + ty, c = c0 + tx;
    tile[ty][tx] = (r < rows && c < cols) ? src[(long long)r * cols + c] : 0.0f;
    __syncthreads();
    const int wc = c0 + ty, wr = r0 + tx;
    if (wc < cols && wr < ldd)
        dst[(long long)wc * ldd + wr] = tile[tx][ty];
}

// Zero-pad rows of length K to length 64: dst[m][0:64] = {src[m][0:K], 0...}.
__global__ void dti_pad64(const float* __restrict__ src, float* __restrict__ dst,
                          int M, int K)
{
    const int i = blockIdx.x * blockDim.x + threadIdx.x;
    if (i >= M * 64) return;
    const int m = i >> 6, k = i & 63;
    dst[i] = (k < K) ? src[m * K + k] : 0.0f;
}

// ---------------------------------------------------------------------------
// GAT attention: e_sym = e + e^T; mask by adj; softmax over axis=1 (the j
// index, for each fixed column k); multiply by adj. One thread per (b,k).
// ---------------------------------------------------------------------------
__global__ void dti_gat_softmax(const float* __restrict__ e,
                                const float* __restrict__ adj,
                                float* __restrict__ att, int N)
{
    const int idx = blockIdx.x * blockDim.x + threadIdx.x;
    if (idx >= DTI_B * N) return;
    const int b = idx / N, k = idx % N;
    const float* eb = e   + (long long)b * N * N;
    const float* ab = adj + (long long)b * N * N;
    float*       ob = att + (long long)b * N * N;

    float m = -3.4e38f;
    for (int j = 0; j < N; ++j) {
        float v = (ab[j * N + k] > 0.0f) ? (eb[j * N + k] + eb[k * N + j]) : -9e15f;
        m = fmaxf(m, v);
    }
    float s = 0.0f;
    for (int j = 0; j < N; ++j) {
        float v = (ab[j * N + k] > 0.0f) ? (eb[j * N + k] + eb[k * N + j]) : -9e15f;
        s += expf(v - m);
    }
    const float inv = 1.0f / s;
    for (int j = 0; j < N; ++j) {
        float a = ab[j * N + k];
        float v = (a > 0.0f) ? (eb[j * N + k] + eb[k * N + j]) : -9e15f;
        ob[j * N + k] = expf(v - m) * inv * a;
    }
}

// ---------------------------------------------------------------------------
// Gate: coeff = sigmoid([x, hp] @ Wg + bg); x = coeff*x + (1-coeff)*hp.
// ---------------------------------------------------------------------------
__global__ void dti_gat_gate(float* __restrict__ x, const float* __restrict__ hp,
                             const float* __restrict__ Wg, const float* __restrict__ bg,
                             int total)
{
    const int row = blockIdx.x * blockDim.x + threadIdx.x;
    if (row >= total) return;
    const float* xr = x  + (long long)row * DTI_D;
    const float* hr = hp + (long long)row * DTI_D;
    float acc = bg[0];
    for (int d = 0; d < DTI_D; ++d) acc += xr[d] * Wg[d];
    for (int d = 0; d < DTI_D; ++d) acc += hr[d] * Wg[DTI_D + d];
    const float c = 1.0f / (1.0f + expf(-acc));
    for (int d = 0; d < DTI_D; ++d)
        x[(long long)row * DTI_D + d] = c * xr[d] + (1.0f - c) * hr[d];
}

__global__ void dti_zero(float* __restrict__ out, int n)
{
    const int i = blockIdx.x * blockDim.x + threadIdx.x;
    if (i < n) out[i] = 0.0f;
}

// ---------------------------------------------------------------------------
// Energy: for each (b, type, n1) block, stream n2; skip pairs with A_int==0.
// hidden = relu(PA1_row + PA2_row); PA1/W2 rows cached in LDS.
// ---------------------------------------------------------------------------
__global__ __launch_bounds__(256)
void dti_energy(const float* __restrict__ PA1, const float* __restrict__ PA2,
                const float* __restrict__ PB1, const float* __restrict__ PB2,
                const float* __restrict__ WA2, const float* __restrict__ bA2,
                const float* __restrict__ WB2, const float* __restrict__ bB2,
                const float* __restrict__ Cc,  const float* __restrict__ A_int,
                const float* __restrict__ dmv, float* __restrict__ out)
{
    const int n1 = blockIdx.x;   // 64
    const int ti = blockIdx.y;   // 7
    const int b  = blockIdx.z;   // 8
    __shared__ float sWA2[DTI_H], sWB2[DTI_H], sPA1[DTI_H], sPB1[DTI_H], sred[256];
    const int tid = threadIdx.x;
    if (tid < DTI_H) {
        sWA2[tid] = WA2[ti * DTI_H + tid];
        sWB2[tid] = WB2[ti * DTI_H + tid];
        sPA1[tid] = PA1[((long long)ti * (DTI_B * DTI_N1) + b * DTI_N1 + n1) * DTI_H + tid];
        sPB1[tid] = PB1[((long long)ti * (DTI_B * DTI_N1) + b * DTI_N1 + n1) * DTI_H + tid];
    }
    __syncthreads();

    const float bcv    = c_BCON[ti];
    const float bc_inv = 1.0f / (3.0f * bcv * bcv);
    const float Cv     = Cc[ti];
    const float biasA  = bA2[ti];
    const float biasB  = bB2[ti];
    const long long aiBase = (((long long)b * DTI_NT + ti) * DTI_N1 + n1) * DTI_N2;

    // Uniform prefetch of the streamed A_int row (global_prefetch_b8 path).
    __builtin_prefetch(&A_int[aiBase + tid], 0, 0);

    float acc = 0.0f;
    for (int n2 = tid; n2 < DTI_N2; n2 += 256) {
        const float ai = A_int[aiBase + n2];
        if (ai > 0.0f) {
            const float4* pa2 = (const float4*)(PA2 +
                ((long long)ti * (DTI_B * DTI_N2) + b * DTI_N2 + n2) * DTI_H);
            const float4* pb2 = (const float4*)(PB2 +
                ((long long)ti * (DTI_B * DTI_N2) + b * DTI_N2 + n2) * DTI_H);
            float sA = 0.0f, sB = 0.0f;
#pragma unroll 4
            for (int h4 = 0; h4 < DTI_H / 4; ++h4) {
                const float4 a4 = pa2[h4];
                const float4 b4 = pb2[h4];
                const int h = h4 * 4;
                sA += fmaxf(sPA1[h+0] + a4.x, 0.f) * sWA2[h+0];
                sA += fmaxf(sPA1[h+1] + a4.y, 0.f) * sWA2[h+1];
                sA += fmaxf(sPA1[h+2] + a4.z, 0.f) * sWA2[h+2];
                sA += fmaxf(sPA1[h+3] + a4.w, 0.f) * sWA2[h+3];
                sB += fmaxf(sPB1[h+0] + b4.x, 0.f) * sWB2[h+0];
                sB += fmaxf(sPB1[h+1] + b4.y, 0.f) * sWB2[h+1];
                sB += fmaxf(sPB1[h+2] + b4.z, 0.f) * sWB2[h+2];
                sB += fmaxf(sPB1[h+3] + b4.w, 0.f) * sWB2[h+3];
            }
            const float Aval = 4.0f / (1.0f + expf(-(sA + biasA)));
            const float Bval = (2.0f * bc_inv) / (1.0f + expf(-(sB + biasB))) + bc_inv;
            const float* dv = dmv + (((long long)b * DTI_N1 + n1) * DTI_N2 + n2) * 3;
            float dm = sqrtf(dv[0]*dv[0] + dv[1]*dv[1] + dv[2]*dv[2] + 1e-10f);
            if (dm < 0.5f) dm = 1e10f;
            const float dd = dm - Cv;
            acc += Aval * (Bval * dd * dd - 1.0f) * ai;
        }
    }
    sred[tid] = acc;
    __syncthreads();
    for (int s = 128; s > 0; s >>= 1) {
        if (tid < s) sred[tid] += sred[tid + s];
        __syncthreads();
    }
    if (tid == 0) atomicAdd(&out[b * DTI_NT + ti], sred[0]);
}

// ---------------------------------------------------------------------------
// Intercept: pooled = sum_n1 h1e*valid; MLP 128->128->1; sigmoid*4; add /7.
// ---------------------------------------------------------------------------
__global__ __launch_bounds__(128)
void dti_intercept(const float* __restrict__ h1e, const float* __restrict__ valid,
                   const float* __restrict__ Wi1, const float* __restrict__ bi1,
                   const float* __restrict__ Wi2, const float* __restrict__ bi2,
                   float* __restrict__ out)
{
    const int b = blockIdx.x;
    const int t = threadIdx.x;
    __shared__ float pooled[DTI_D], hid[DTI_D];
    float s = 0.0f;
    for (int n = 0; n < DTI_N1; ++n)
        s += h1e[((long long)b * DTI_N1 + n) * DTI_D + t] * valid[b * DTI_N1 + n];
    pooled[t] = s;
    __syncthreads();
    float hv = bi1[t];
    for (int d = 0; d < DTI_D; ++d) hv += pooled[d] * Wi1[d * DTI_D + t];
    hid[t] = fmaxf(hv, 0.0f) * Wi2[t];
    __syncthreads();
    for (int st = 64; st > 0; st >>= 1) {
        if (t < st) hid[t] += hid[t + st];
        __syncthreads();
    }
    if (t == 0) {
        const float ic = 4.0f / (1.0f + expf(-(hid[0] + bi2[0])));
        for (int i = 0; i < DTI_NT; ++i)
            atomicAdd(&out[b * DTI_NT + i], ic / (float)DTI_NT);
    }
}

// ---------------------------------------------------------------------------
extern "C" void kernel_launch(void* const* d_in, const int* in_sizes, int n_in,
                              void* d_out, int out_size, void* d_ws, size_t ws_size,
                              hipStream_t stream)
{
    const float* h1     = (const float*)d_in[0];
    const float* adj1   = (const float*)d_in[1];
    const float* h2     = (const float*)d_in[2];
    const float* adj2   = (const float*)d_in[3];
    const float* A_int  = (const float*)d_in[4];
    const float* dmv    = (const float*)d_in[5];
    const float* valid  = (const float*)d_in[6];
    const float* W_emb  = (const float*)d_in[7];
    const float* gW     = (const float*)d_in[8];
    const float* gWb    = (const float*)d_in[9];
    const float* gA     = (const float*)d_in[10];
    const float* gGateW = (const float*)d_in[11];
    const float* gGateb = (const float*)d_in[12];
    const float* WA1    = (const float*)d_in[13];
    const float* bA1    = (const float*)d_in[14];
    const float* WA2    = (const float*)d_in[15];
    const float* bA2    = (const float*)d_in[16];
    const float* WB1    = (const float*)d_in[17];
    const float* bB1    = (const float*)d_in[18];
    const float* WB2    = (const float*)d_in[19];
    const float* bB2    = (const float*)d_in[20];
    const float* Cc     = (const float*)d_in[21];
    const float* Wi1    = (const float*)d_in[22];
    const float* bi1    = (const float*)d_in[23];
    const float* Wi2    = (const float*)d_in[24];
    const float* bi2    = (const float*)d_in[25];
    float* out = (float*)d_out;

    // Workspace layout (floats). Total ~16.0M floats = ~64 MB.
    float* ws    = (float*)d_ws;
    float* h1e   = ws;                                     // 8*64*128
    float* h2e   = h1e  + DTI_B * DTI_N1 * DTI_D;          // 8*512*128
    float* t_h   = h2e  + DTI_B * DTI_N2 * DTI_D;
    float* t_hA  = t_h  + DTI_B * DTI_N2 * DTI_D;
    float* e_b   = t_hA + DTI_B * DTI_N2 * DTI_D;          // 8*512*512
    float* att   = e_b  + DTI_B * DTI_N2 * DTI_N2;
    float* hp    = att  + DTI_B * DTI_N2 * DTI_N2;
    float* PA1   = hp   + DTI_B * DTI_N2 * DTI_D;          // 7*512*128
    float* PB1   = PA1  + DTI_NT * DTI_B * DTI_N1 * DTI_H;
    float* PA2   = PB1  + DTI_NT * DTI_B * DTI_N1 * DTI_H; // 7*4096*128
    float* PB2   = PA2  + DTI_NT * DTI_B * DTI_N2 * DTI_H;
    float* h1p   = PB2  + DTI_NT * DTI_B * DTI_N2 * DTI_H; // 512*64
    float* h2p   = h1p  + DTI_B * DTI_N1 * 64;             // 4096*64
    float* WeT   = h2p  + DTI_B * DTI_N2 * 64;             // 128*64
    float* gWT   = WeT  + DTI_D * 64;                      // 3*128*128
    float* gAT   = gWT  + 3 * DTI_D * DTI_D;               // 3*128*128
    float* t_hT  = gAT  + 3 * DTI_D * DTI_D;               // 8*128*512 (max)
    float* WA1Tt = t_hT + DTI_B * DTI_D * DTI_N2;          // 7*128*128 each
    float* WA1Tb = WA1Tt + DTI_NT * DTI_H * DTI_D;
    float* WB1Tt = WA1Tb + DTI_NT * DTI_H * DTI_D;
    float* WB1Tb = WB1Tt + DTI_NT * DTI_H * DTI_D;

    const dim3 wave(32);
    const dim3 tb(256);

    // --- Pre-transpose all weight B-operands into [N][K] layout ---
    dti_transpose<<<dim3(4, 8, 1), tb, 0, stream>>>(W_emb, WeT, 56, DTI_D, 64, 0, 0);
    dti_transpose<<<dim3(8, 8, 3), tb, 0, stream>>>(gW, gWT, DTI_D, DTI_D, DTI_D,
                                                    DTI_D*DTI_D, DTI_D*DTI_D);
    dti_transpose<<<dim3(8, 8, 3), tb, 0, stream>>>(gA, gAT, DTI_D, DTI_D, DTI_D,
                                                    DTI_D*DTI_D, DTI_D*DTI_D);
    dti_transpose<<<dim3(8, 8, DTI_NT), tb, 0, stream>>>(WA1, WA1Tt, DTI_D, DTI_H, DTI_D,
                                                         2*DTI_D*DTI_H, DTI_H*DTI_D);
    dti_transpose<<<dim3(8, 8, DTI_NT), tb, 0, stream>>>(WA1 + DTI_D*DTI_H, WA1Tb, DTI_D, DTI_H, DTI_D,
                                                         2*DTI_D*DTI_H, DTI_H*DTI_D);
    dti_transpose<<<dim3(8, 8, DTI_NT), tb, 0, stream>>>(WB1, WB1Tt, DTI_D, DTI_H, DTI_D,
                                                         2*DTI_D*DTI_H, DTI_H*DTI_D);
    dti_transpose<<<dim3(8, 8, DTI_NT), tb, 0, stream>>>(WB1 + DTI_D*DTI_H, WB1Tb, DTI_D, DTI_H, DTI_D,
                                                         2*DTI_D*DTI_H, DTI_H*DTI_D);

    // --- Pad embed inputs to K=64 and run embedding GEMMs (cols 128 = 2x64) ---
    dti_pad64<<<dim3((DTI_B*DTI_N1*64 + 255)/256), tb, 0, stream>>>(h1, h1p, DTI_B*DTI_N1, 56);
    dti_pad64<<<dim3((DTI_B*DTI_N2*64 + 255)/256), tb, 0, stream>>>(h2, h2p, DTI_B*DTI_N2, 56);
    dti_gemm_wmma<<<dim3(DTI_B*DTI_N1/16, DTI_D/64, 1), wave, 0, stream>>>(
        h1p, WeT, h1e, nullptr, 64, 64, 64, DTI_D, 0, 0, 0, 0, 0);
    dti_gemm_wmma<<<dim3(DTI_B*DTI_N2/16, DTI_D/64, 1), wave, 0, stream>>>(
        h2p, WeT, h2e, nullptr, 64, 64, 64, DTI_D, 0, 0, 0, 0, 0);

    // --- 3 GAT layers on each graph ---
    for (int l = 0; l < 3; ++l) {
        const float* WlT = gWT    + (long long)l * DTI_D * DTI_D;
        const float* Wbl = gWb    + (long long)l * DTI_D;
        const float* AlT = gAT    + (long long)l * DTI_D * DTI_D;
        const float* Wgl = gGateW + (long long)l * 2 * DTI_D;
        const float* bgl = gGateb + l;
        for (int g = 0; g < 2; ++g) {
            float*       he  = g ? h2e  : h1e;
            const int    N   = g ? DTI_N2 : DTI_N1;
            const float* adj = g ? adj2 : adj1;
            const int    MT  = DTI_B * N;
            // h = x@W + Wb
            dti_gemm_wmma<<<dim3(MT/16, DTI_D/64, 1), wave, 0, stream>>>(
                he, WlT, t_h, Wbl, DTI_D, DTI_D, DTI_D, DTI_D, 0, 0, 0, 0, 0);
            // hA = h@A
            dti_gemm_wmma<<<dim3(MT/16, DTI_D/64, 1), wave, 0, stream>>>(
                t_h, AlT, t_hA, nullptr, DTI_D, DTI_D, DTI_D, DTI_D, 0, 0, 0, 0, 0);
            // e[b] = hA[b] @ h[b]^T   (B already [N][K] -> use t_h directly)
            dti_gemm_wmma<<<dim3(N/16, N/64, DTI_B), wave, 0, stream>>>(
                t_hA, t_h, e_b, nullptr, DTI_D, DTI_D, DTI_D, N, 0,
                (long long)N*DTI_D, (long long)N*DTI_D, (long long)N*N, 0);
            // att = softmax(mask(e+e^T), axis=1) * adj
            dti_gat_softmax<<<dim3((DTI_B*N + 127)/128), dim3(128), 0, stream>>>(
                e_b, adj, att, N);
            // t_hT[b] = t_h[b]^T  ([128][N]) for the hp GEMM
            dti_transpose<<<dim3(N/16, DTI_D/16, DTI_B), tb, 0, stream>>>(
                t_h, t_hT, N, DTI_D, N, (long long)N*DTI_D, (long long)DTI_D*N);
            // hp[b] = relu(att[b] @ t_h[b])
            dti_gemm_wmma<<<dim3(N/16, DTI_D/64, DTI_B), wave, 0, stream>>>(
                att, t_hT, hp, nullptr, N, N, N, DTI_D, 1,
                (long long)N*N, (long long)DTI_D*N, (long long)N*DTI_D, 0);
            // gated residual update (in place on he)
            dti_gat_gate<<<dim3((MT + 127)/128), dim3(128), 0, stream>>>(
                he, hp, Wgl, bgl, MT);
        }
    }

    // --- Factorized pair-MLP first layers, batched over 7 types (grid.z) ---
    dti_gemm_wmma<<<dim3(DTI_B*DTI_N1/16, DTI_H/64, DTI_NT), wave, 0, stream>>>(
        h1e, WA1Tt, PA1, bA1, DTI_D, DTI_D, DTI_D, DTI_H, 0,
        0, (long long)DTI_H*DTI_D, (long long)DTI_B*DTI_N1*DTI_H, DTI_H);
    dti_gemm_wmma<<<dim3(DTI_B*DTI_N2/16, DTI_H/64, DTI_NT), wave, 0, stream>>>(
        h2e, WA1Tb, PA2, nullptr, DTI_D, DTI_D, DTI_D, DTI_H, 0,
        0, (long long)DTI_H*DTI_D, (long long)DTI_B*DTI_N2*DTI_H, 0);
    dti_gemm_wmma<<<dim3(DTI_B*DTI_N1/16, DTI_H/64, DTI_NT), wave, 0, stream>>>(
        h1e, WB1Tt, PB1, bB1, DTI_D, DTI_D, DTI_D, DTI_H, 0,
        0, (long long)DTI_H*DTI_D, (long long)DTI_B*DTI_N1*DTI_H, DTI_H);
    dti_gemm_wmma<<<dim3(DTI_B*DTI_N2/16, DTI_H/64, DTI_NT), wave, 0, stream>>>(
        h2e, WB1Tb, PB2, nullptr, DTI_D, DTI_D, DTI_D, DTI_H, 0,
        0, (long long)DTI_H*DTI_D, (long long)DTI_B*DTI_N2*DTI_H, 0);

    // --- Energies + intercept ---
    dti_zero<<<dim3(1), dim3(64), 0, stream>>>(out, DTI_B * DTI_NT);
    dti_energy<<<dim3(DTI_N1, DTI_NT, DTI_B), dim3(256), 0, stream>>>(
        PA1, PA2, PB1, PB2, WA2, bA2, WB2, bB2, Cc, A_int, dmv, out);
    dti_intercept<<<dim3(DTI_B), dim3(128), 0, stream>>>(
        h1e, valid, Wi1, bi1, Wi2, bi2, out);

    (void)in_sizes; (void)n_in; (void)out_size; (void)ws_size;
}
